// CARAFE_7078106104459
// MI455X (gfx1250) — compile-verified, matrix operand order checked by reading, not run on hardware
//
#include <hip/hip_runtime.h>
#include <hip/hip_bf16.h>

// ---------------------------------------------------------------------------
// CARAFE + head for MI455X (gfx1250), wave32, bf16 WMMA 16x16x32 (f32 acc).
// Shapes fixed by reference: N=16, C=256, H=W=48, UP=2, K=3, P48=36864,
// P96=147456 pixels after upsample.
// ---------------------------------------------------------------------------

typedef unsigned short u16;
typedef __attribute__((ext_vector_type(16))) __bf16  v16bf;
typedef __attribute__((ext_vector_type(8)))  float   v8f;
typedef __attribute__((ext_vector_type(8)))  unsigned int v8u;

__device__ __forceinline__ u16 f2bf(float f) {
  unsigned int u = __builtin_bit_cast(unsigned int, f);
  unsigned int r = u + 0x7FFFu + ((u >> 16) & 1u);   // RNE
  return (u16)(r >> 16);
}
__device__ __forceinline__ float bf2f(u16 h) {
  unsigned int u = ((unsigned int)h) << 16;
  return __builtin_bit_cast(float, u);
}
__device__ __forceinline__ float sigm(float x) { return 1.f / (1.f + __expf(-x)); }

// ---------------------------------------------------------------------------
// 1) NCHW fp32 -> pixel-major bf16  inT[(n,h,w)][c]
//    grid (2, 8, N*H), block (32,8); LDS tile transpose, coalesced both ways.
// ---------------------------------------------------------------------------
__global__ __launch_bounds__(256) void transpose_kernel(
    const float* __restrict__ in, u16* __restrict__ inT)
{
  __shared__ float tile[32][33];
  const int tx = threadIdx.x, ty = threadIdx.y;
  const int wBase = blockIdx.x * 32, cBase = blockIdx.y * 32;
  const int nh = blockIdx.z, n = nh / 48, h = nh % 48;
  const int w = wBase + tx;
  if (w < 48) {
    for (int cy = ty; cy < 32; cy += 8) {
      int c = cBase + cy;
      tile[cy][tx] = in[((size_t)(n * 256 + c) * 48 + h) * 48 + w];
    }
  }
  __syncthreads();
  for (int wy = ty; wy < 32; wy += 8) {
    int ww = wBase + wy;
    if (ww < 48) {
      int c = cBase + tx;
      inT[((size_t)((n * 48 + h) * 48 + ww)) * 256 + c] = f2bf(tile[tx][wy]);
    }
  }
}

// ---------------------------------------------------------------------------
// 2a) Fold out1 -> outc1 -> BN1 into one 256x256 bf16 matrix + fp32 bias.
//     grid 256 (o), block 256 (i).
// ---------------------------------------------------------------------------
__global__ __launch_bounds__(256) void fold_w1_kernel(
    const float* __restrict__ oc1w, const float* __restrict__ o1w,
    const float* __restrict__ o1b,  const float* __restrict__ oc1b,
    const float* __restrict__ g, const float* __restrict__ bb,
    const float* __restrict__ m, const float* __restrict__ v,
    u16* __restrict__ W1eff, float* __restrict__ b1eff)
{
  const int o = blockIdx.x, i = threadIdx.x;
  const float s = g[o] * rsqrtf(v[o] + 1e-5f);
  float acc = 0.f;
  for (int j = 0; j < 256; ++j)
    acc = fmaf(oc1w[o * 256 + j], o1w[j * 256 + i], acc);
  W1eff[o * 256 + i] = f2bf(s * acc);
  if (i == 0) {
    float ba = oc1b[o];
    for (int j = 0; j < 256; ++j) ba = fmaf(oc1w[o * 256 + j], o1b[j], ba);
    b1eff[o] = s * ba + bb[o] - m[o] * s;
  }
}

// 2b) Fold BN2 into outc2; bf16 copy of down_w.
__global__ void prep_small_kernel(
    const float* __restrict__ oc2w, const float* __restrict__ oc2b,
    const float* __restrict__ g, const float* __restrict__ bb,
    const float* __restrict__ m, const float* __restrict__ v,
    const float* __restrict__ dw,
    u16* __restrict__ W2eff, float* __restrict__ b2eff, u16* __restrict__ downW)
{
  const int idx = blockIdx.x * 256 + threadIdx.x;
  if (idx < 65536) {
    const int o = idx >> 8;
    const float s = g[o] * rsqrtf(v[o] + 1e-5f);
    W2eff[idx] = f2bf(s * oc2w[idx]);
    if ((idx & 255) == 0) b2eff[o] = s * oc2b[o] + bb[o] - m[o] * s;
  } else if (idx < 65536 + 16384) {
    const int j = idx - 65536;
    downW[j] = f2bf(dw[j]);
  }
}

// ---------------------------------------------------------------------------
// 3) bf16 WMMA GEMM:  out[P x nOut] = act(A[P x 256] @ W[nOut x 256]^T + bias)
//    Block = 16 rows x 256 cols (8 waves, 1 wave = 16x32 via 2 WMMA tiles).
//    A tile staged in LDS (padded rows, ds_load_b128 frag reads).
//    Frag layout per CDNA5 ISA 7.12.2: lane L holds row/col (L&15),
//    K runs [klo,klo+8) and [klo+16,klo+24), klo = (L<16)?0:8.
//    MODE 0: fp32 row-major, no relu   (down conv, nOut=64)
//    MODE 1: bf16 row-major, relu      (fused out1+outc1+BN1)
//    MODE 2: fp32 NCHW into d_out, relu (outc2+BN2)
// ---------------------------------------------------------------------------
__device__ __forceinline__ v16bf load_frag(const u16* __restrict__ p) {
  uint4 lo = *reinterpret_cast<const uint4*>(p);
  uint4 hi = *reinterpret_cast<const uint4*>(p + 16);
  v8u u;
  u[0] = lo.x; u[1] = lo.y; u[2] = lo.z; u[3] = lo.w;
  u[4] = hi.x; u[5] = hi.y; u[6] = hi.z; u[7] = hi.w;
  return __builtin_bit_cast(v16bf, u);
}

template <int MODE>
__global__ __launch_bounds__(256) void gemm16_bf16(
    const u16* __restrict__ A, const u16* __restrict__ W,
    const float* __restrict__ bias, void* __restrict__ out, int nOut)
{
  __shared__ u16 As[16][256 + 8];               // +8 u16 pad: spread LDS banks
  const int tid = threadIdx.x;
  const int rowBase = blockIdx.x * 16;

  // cooperative A-tile load: 16 rows x 256 bf16 = 8 KB, 16B chunks
  {
    const uint4* src = reinterpret_cast<const uint4*>(A + (size_t)rowBase * 256);
    for (int i = tid; i < 16 * 32; i += 256) {
      const int r = i >> 5, ck = i & 31;
      *reinterpret_cast<uint4*>(&As[r][ck * 8]) = src[r * 32 + ck];
    }
  }
  __syncthreads();

  const int wave = tid >> 5;
  const int lane = tid & 31;
  const int colBase = wave * 32;
  if (colBase >= nOut) return;                  // wave-uniform; EXEC stays full

  const int rsel = lane & 15;
  const int ksel = (lane >> 4) << 3;            // 0 or 8

  v8f acc0 = {0.f, 0.f, 0.f, 0.f, 0.f, 0.f, 0.f, 0.f};
  v8f acc1 = {0.f, 0.f, 0.f, 0.f, 0.f, 0.f, 0.f, 0.f};

  const u16* bRow0 = W + (size_t)(colBase + rsel) * 256 + ksel;
  const u16* bRow1 = bRow0 + (size_t)16 * 256;
  const u16* aRow  = &As[rsel][ksel];

  #pragma unroll
  for (int kb = 0; kb < 256; kb += 32) {
    v16bf aF  = load_frag(aRow + kb);           // ds_load_b128 x2
    v16bf bF0 = load_frag(bRow0 + kb);          // global_load_b128 x2
    v16bf bF1 = load_frag(bRow1 + kb);
    acc0 = __builtin_amdgcn_wmma_f32_16x16x32_bf16(
        false, aF, false, bF0, (short)0, acc0, false, false);
    acc1 = __builtin_amdgcn_wmma_f32_16x16x32_bf16(
        false, aF, false, bF1, (short)0, acc1, false, false);
  }

  // epilogue: element j of acc -> row (j + 8*(lane>=16)), col (lane&15)
  const int rowHi = (lane >> 4) << 3;
  const int c0 = colBase + rsel, c1 = c0 + 16;
  const float b0 = bias[c0], b1 = bias[c1];
  #pragma unroll
  for (int j = 0; j < 8; ++j) {
    const int r = rowBase + j + rowHi;
    float v0 = acc0[j] + b0;
    float v1 = acc1[j] + b1;
    if (MODE >= 1) { v0 = fmaxf(v0, 0.f); v1 = fmaxf(v1, 0.f); }
    if (MODE == 0) {
      float* o = (float*)out;
      o[(size_t)r * nOut + c0] = v0;
      o[(size_t)r * nOut + c1] = v1;
    } else if (MODE == 1) {
      u16* o = (u16*)out;
      o[(size_t)r * 256 + c0] = f2bf(v0);
      o[(size_t)r * 256 + c1] = f2bf(v1);
    } else {
      float* o = (float*)out;                    // NCHW (16,256,96,96)
      const int n = r / 9216, rem = r % 9216, y = rem / 96, xw = rem % 96;
      o[(((size_t)n * 256 + c0) * 96 + y) * 96 + xw] = v0;
      o[(((size_t)n * 256 + c1) * 96 + y) * 96 + xw] = v1;
    }
  }
}

// ---------------------------------------------------------------------------
// 4) enc 3x3 conv (64 -> 36) + per-(a,b) softmax over 9 taps.
//    1 thread per base pixel; masks[p][ab][q].
// ---------------------------------------------------------------------------
__global__ __launch_bounds__(256) void enc_softmax_kernel(
    const float* __restrict__ mid, const float* __restrict__ ew,
    const float* __restrict__ eb, float* __restrict__ masks)
{
  const int p = blockIdx.x * 256 + threadIdx.x;
  if (p >= 36864) return;
  const int n = p / 2304, rem = p % 2304, h = rem / 48, w = rem % 48;

  float acc[36];
  #pragma unroll
  for (int e = 0; e < 36; ++e) acc[e] = eb[e];

  for (int dy = 0; dy < 3; ++dy) {
    const int hh = h + dy - 1;
    if (hh < 0 || hh >= 48) continue;
    for (int dx = 0; dx < 3; ++dx) {
      const int ww = w + dx - 1;
      if (ww < 0 || ww >= 48) continue;
      const float* mrow = mid + ((size_t)((n * 48 + hh) * 48 + ww)) * 64;
      const int wo = dy * 3 + dx;
      for (int c = 0; c < 64; ++c) {
        const float mv = mrow[c];
        #pragma unroll
        for (int e = 0; e < 36; ++e)
          acc[e] = fmaf(mv, ew[(e * 64 + c) * 9 + wo], acc[e]);
      }
    }
  }

  float* mo = masks + (size_t)p * 36;
  #pragma unroll
  for (int ab = 0; ab < 4; ++ab) {               // channel e = q*4 + ab
    float mx = -1e30f;
    #pragma unroll
    for (int q = 0; q < 9; ++q) mx = fmaxf(mx, acc[q * 4 + ab]);
    float s = 0.f, ex[9];
    #pragma unroll
    for (int q = 0; q < 9; ++q) { ex[q] = __expf(acc[q * 4 + ab] - mx); s += ex[q]; }
    const float inv = 1.f / s;
    #pragma unroll
    for (int q = 0; q < 9; ++q) mo[ab * 9 + q] = ex[q] * inv;
  }
}

// ---------------------------------------------------------------------------
// 5) Content-aware reassembly: block per base pixel, thread per channel.
//    up[(n, 2h+a, 2w+b)][c] = sum_q in[n,c,h+dy,w+dx] * mask[ab][q]
// ---------------------------------------------------------------------------
__global__ __launch_bounds__(256) void reassemble_kernel(
    const u16* __restrict__ inT, const float* __restrict__ masks,
    u16* __restrict__ up)
{
  __shared__ float lm[36];
  const int p = blockIdx.x, c = threadIdx.x;
  if (c < 36) lm[c] = masks[(size_t)p * 36 + c];
  __syncthreads();

  const int n = p / 2304, rem = p % 2304, h = rem / 48, w = rem % 48;
  float a0 = 0.f, a1 = 0.f, a2 = 0.f, a3 = 0.f;
  #pragma unroll
  for (int q = 0; q < 9; ++q) {
    const int hh = h + q / 3 - 1, ww = w + q % 3 - 1;
    float v = 0.f;
    if (hh >= 0 && hh < 48 && ww >= 0 && ww < 48)
      v = bf2f(inT[((size_t)((n * 48 + hh) * 48 + ww)) * 256 + c]);
    a0 = fmaf(v, lm[0 * 9 + q], a0);
    a1 = fmaf(v, lm[1 * 9 + q], a1);
    a2 = fmaf(v, lm[2 * 9 + q], a2);
    a3 = fmaf(v, lm[3 * 9 + q], a3);
  }
  const size_t base = ((size_t)(n * 96 + 2 * h) * 96 + 2 * w);
  up[(base)      * 256 + c] = f2bf(a0);          // (a=0,b=0)
  up[(base + 1)  * 256 + c] = f2bf(a1);          // (a=0,b=1)
  up[(base + 96) * 256 + c] = f2bf(a2);          // (a=1,b=0)
  up[(base + 97) * 256 + c] = f2bf(a3);          // (a=1,b=1)
}

// ---------------------------------------------------------------------------
// 6) SCSE gating on x (fp32 NCHW in d_out)
// ---------------------------------------------------------------------------
__global__ __launch_bounds__(256) void mean_kernel(
    const float* __restrict__ x, float* __restrict__ mean)
{
  __shared__ float red[256];
  const int b = blockIdx.x;                      // b = n*256 + c
  const float* base = x + (size_t)b * 9216;
  float s = 0.f;
  for (int i = threadIdx.x; i < 9216; i += 256) s += base[i];
  red[threadIdx.x] = s;
  __syncthreads();
  for (int off = 128; off > 0; off >>= 1) {
    if (threadIdx.x < off) red[threadIdx.x] += red[threadIdx.x + off];
    __syncthreads();
  }
  if (threadIdx.x == 0) mean[b] = red[0] * (1.f / 9216.f);
}

__global__ __launch_bounds__(256) void gate_kernel(
    const float* __restrict__ mean,
    const float* __restrict__ w1, const float* __restrict__ b1,
    const float* __restrict__ w2, const float* __restrict__ b2,
    float* __restrict__ chanGate)
{
  __shared__ float sm[256];
  __shared__ float s1[16];
  const int n = blockIdx.x, t = threadIdx.x;
  sm[t] = mean[n * 256 + t];
  __syncthreads();
  if (t < 16) {
    float a = b1[t];
    for (int c = 0; c < 256; ++c) a = fmaf(w1[t * 256 + c], sm[c], a);
    s1[t] = fmaxf(a, 0.f);
  }
  __syncthreads();
  float a = b2[t];
  #pragma unroll
  for (int j = 0; j < 16; ++j) a = fmaf(w2[t * 16 + j], s1[j], a);
  chanGate[n * 256 + t] = sigm(a);
}

__global__ __launch_bounds__(96) void pixgate_kernel(
    const float* __restrict__ x, const float* __restrict__ cgw,
    const float* __restrict__ cgb, float* __restrict__ pg)
{
  const int n = blockIdx.x / 96, y = blockIdx.x % 96, xw = threadIdx.x;
  float a = cgb[0];
  for (int c = 0; c < 256; ++c)
    a = fmaf(cgw[c], x[(((size_t)n * 256 + c) * 96 + y) * 96 + xw], a);
  pg[((size_t)n * 96 + y) * 96 + xw] = sigm(a);
}

__global__ __launch_bounds__(256) void final_kernel(
    float* __restrict__ x, const float* __restrict__ chanGate,
    const float* __restrict__ pg)
{
  const size_t idx = (size_t)blockIdx.x * 256 + threadIdx.x;
  if (idx >= (size_t)37748736) return;
  const int n = (int)(idx / (9216 * 256));
  const int c = (int)((idx / 9216) & 255);
  const int pix = (int)(idx % 9216);
  x[idx] = x[idx] * (chanGate[n * 256 + c] + pg[(size_t)n * 9216 + pix]);
}

// ---------------------------------------------------------------------------
// Launch. Workspace budget ~186 MB (offsets all 256B-aligned).
// ---------------------------------------------------------------------------
extern "C" void kernel_launch(void* const* d_in, const int* in_sizes, int n_in,
                              void* d_out, int out_size, void* d_ws, size_t ws_size,
                              hipStream_t stream)
{
  (void)in_sizes; (void)n_in; (void)out_size; (void)ws_size;
  const float* in_t   = (const float*)d_in[0];
  const float* down_w = (const float*)d_in[1];
  const float* down_b = (const float*)d_in[2];
  const float* enc_w  = (const float*)d_in[3];
  const float* enc_b  = (const float*)d_in[4];
  const float* out1_w = (const float*)d_in[5];
  const float* out1_b = (const float*)d_in[6];
  const float* oc1_w  = (const float*)d_in[7];
  const float* oc1_b  = (const float*)d_in[8];
  const float* bn1_g  = (const float*)d_in[9];
  const float* bn1_b  = (const float*)d_in[10];
  const float* bn1_m  = (const float*)d_in[11];
  const float* bn1_v  = (const float*)d_in[12];
  const float* oc2_w  = (const float*)d_in[13];
  const float* oc2_b  = (const float*)d_in[14];
  const float* bn2_g  = (const float*)d_in[15];
  const float* bn2_b  = (const float*)d_in[16];
  const float* bn2_m  = (const float*)d_in[17];
  const float* bn2_v  = (const float*)d_in[18];
  const float* sg_w1  = (const float*)d_in[19];
  const float* sg_b1  = (const float*)d_in[20];
  const float* sg_w2  = (const float*)d_in[21];
  const float* sg_b2  = (const float*)d_in[22];
  const float* cg_w   = (const float*)d_in[23];
  const float* cg_b   = (const float*)d_in[24];

  char* ws = (char*)d_ws;
  u16*   inT    = (u16*)  (ws + 0);             //  18,874,368 B
  float* mid    = (float*)(ws + 18874368);      //   9,437,184 B
  float* masks  = (float*)(ws + 28311552);      //   5,308,416 B
  u16*   up     = (u16*)  (ws + 33619968);      //  75,497,472 B
  u16*   x1     = (u16*)  (ws + 109117440);     //  75,497,472 B
  u16*   W1eff  = (u16*)  (ws + 184614912);     //     131,072 B
  float* b1eff  = (float*)(ws + 184745984);     //       1,024 B
  u16*   W2eff  = (u16*)  (ws + 184747008);     //     131,072 B
  float* b2eff  = (float*)(ws + 184878080);     //       1,024 B
  u16*   downW  = (u16*)  (ws + 184879104);     //      32,768 B
  float* meanb  = (float*)(ws + 184911872);     //      16,384 B
  float* chanG  = (float*)(ws + 184928256);     //      16,384 B
  float* pg     = (float*)(ws + 184944640);     //     589,824 B

  // 1) transpose + bf16
  transpose_kernel<<<dim3(2, 8, 768), dim3(32, 8), 0, stream>>>(in_t, inT);
  // 2) weight folding
  fold_w1_kernel<<<256, 256, 0, stream>>>(oc1_w, out1_w, out1_b, oc1_b,
                                          bn1_g, bn1_b, bn1_m, bn1_v,
                                          W1eff, b1eff);
  prep_small_kernel<<<(81920 + 255) / 256, 256, 0, stream>>>(
      oc2_w, oc2_b, bn2_g, bn2_b, bn2_m, bn2_v, down_w, W2eff, b2eff, downW);
  // 3) down 1x1 conv (WMMA GEMM, nOut=64)
  gemm16_bf16<0><<<36864 / 16, 256, 0, stream>>>(inT, downW, down_b, mid, 64);
  // 4) enc 3x3 + softmax
  enc_softmax_kernel<<<144, 256, 0, stream>>>(mid, enc_w, enc_b, masks);
  // 5) reassembly -> up (bf16 pixel-major)
  reassemble_kernel<<<36864, 256, 0, stream>>>(inT, masks, up);
  // 6) fused out1+outc1+BN1+ReLU (WMMA GEMM)
  gemm16_bf16<1><<<147456 / 16, 256, 0, stream>>>(up, W1eff, b1eff, x1, 256);
  // 7) outc2+BN2+ReLU -> fp32 NCHW in d_out (WMMA GEMM)
  gemm16_bf16<2><<<147456 / 16, 256, 0, stream>>>(x1, W2eff, b2eff, d_out, 256);
  // 8) SCSE gates, final scale in place on d_out
  mean_kernel<<<4096, 256, 0, stream>>>((const float*)d_out, meanb);
  gate_kernel<<<16, 256, 0, stream>>>(meanb, sg_w1, sg_b1, sg_w2, sg_b2, chanG);
  pixgate_kernel<<<1536, 96, 0, stream>>>((const float*)d_out, cg_w, cg_b, pg);
  final_kernel<<<147456, 256, 0, stream>>>((float*)d_out, chanG, pg);
}
